// ParallelBackbone_12695923327030
// MI455X (gfx1250) — compile-verified, matrix operand order checked by reading, not run on hardware
//
#include <hip/hip_runtime.h>
#include <math.h>

// ---------------- problem constants (from reference) ----------------
#define Bb 128
#define Pp 128
#define Kk 16
#define Hh 8
#define IND 7
#define IDD 64
#define CTX 128
#define NOD 32
#define EDG 8
#define MSG 64
#define EPSF 1e-8f

#define WAVES 4
#define NITER 8   // nodes per wave -> 32 nodes per block

typedef __attribute__((ext_vector_type(16))) __bf16 v16bf;
typedef __attribute__((ext_vector_type(8)))  float  v8f;

__host__ __device__ constexpr int cm_of(int cin)  { return 2 * cin + EDG; }
__host__ __device__ constexpr int cmp_of(int cin) { return ((cm_of(cin) + 31) / 32) * 32; }
__host__ __device__ constexpr size_t perwave_bytes(int cin) {
  size_t s = (size_t)16 * cmp_of(cin) * 2   // Xh (bf16)
           + (size_t)MSG * 16 * 4           // mmv
           + (size_t)Hh * 16 * 4            // lg
           + 192 * 4                        // catv
           + (size_t)cin * 4                // xi
           + 16 * 4 + 16 * 4;               // sj, sn
  return (s + 255) & ~(size_t)255;
}
__host__ __device__ constexpr size_t smem_bytes(int cin) {
  return (size_t)MSG * cmp_of(cin) * 2 + (size_t)WAVES * perwave_bytes(cin);
}

__device__ inline unsigned pack_bf(float a, float b) {
  union { unsigned u; __bf16 h[2]; } z;
  z.h[0] = (__bf16)a; z.h[1] = (__bf16)b;
  return z.u;
}

// ---------------- identity branch: writes d_out[:,0:64,:] ----------------
__global__ void identity_kernel(const float* __restrict__ feat,
                                const unsigned char* __restrict__ mask,
                                const float* __restrict__ s0, const float* __restrict__ b0,
                                const float* __restrict__ W,
                                const float* __restrict__ s1, const float* __restrict__ b1,
                                float* __restrict__ out) {
  int t = blockIdx.x * blockDim.x + threadIdx.x;
  if (t >= Bb * Pp) return;
  int b = t / Pp, p = t % Pp;
  float f[IND];
#pragma unroll
  for (int c = 0; c < IND; ++c) f[c] = feat[(b * IND + c) * Pp + p] * s0[c] + b0[c];
  float mf = mask[b * Pp + p] ? 1.f : 0.f;
  for (int o = 0; o < IDD; ++o) {
    float v = 0.f;
#pragma unroll
    for (int c = 0; c < IND; ++c) v += W[o * IND + c] * f[c];
    v = v * s1[o] + b1[o];
    out[(b * (IDD + CTX) + o) * Pp + p] = (v > 0.f ? v : 0.f) * mf;
  }
}

// ---------------- kNN (stable k-smallest, drop self) ----------------
__global__ void knn_kernel(const float* __restrict__ pts,
                           const unsigned char* __restrict__ mask,
                           int* __restrict__ idx) {
  int t = blockIdx.x * blockDim.x + threadIdx.x;
  if (t >= Bb * Pp) return;
  int b = t / Pp, p = t % Pp;
  const float* eta = &pts[(b * 2 + 0) * Pp];
  const float* phi = &pts[(b * 2 + 1) * Pp];
  const unsigned char* mk = &mask[b * Pp];
  float e0 = mk[p] ? eta[p] : 1e9f;
  float f0 = mk[p] ? phi[p] : 1e9f;
  float bd[Kk + 1];
  int bi[Kk + 1];
#pragma unroll
  for (int i = 0; i <= Kk; ++i) { bd[i] = 3.4e38f; bi[i] = 0; }
  for (int q = 0; q < Pp; ++q) {
    float eq = mk[q] ? eta[q] : 1e9f;
    float fq = mk[q] ? phi[q] : 1e9f;
    float de = e0 - eq, df = f0 - fq;
    float d2 = de * de + df * df;
    if (d2 < bd[Kk]) {
      int i = Kk;
      while (i > 0 && d2 < bd[i - 1]) { bd[i] = bd[i - 1]; bi[i] = bi[i - 1]; --i; }
      bd[i] = d2; bi[i] = q;
    }
  }
#pragma unroll
  for (int k = 0; k < Kk; ++k) idx[t * Kk + k] = bi[k + 1];
}

// -------- pairwise Lorentz features -> edge embed e (bf16, layer-invariant) --------
__global__ void edge_kernel(const float* __restrict__ lv,
                            const unsigned char* __restrict__ mask,
                            const int* __restrict__ idx,
                            const float* __restrict__ ebs, const float* __restrict__ ebb,
                            const float* __restrict__ edge_W,
                            unsigned short* __restrict__ e8, int* __restrict__ nulle) {
  int t = blockIdx.x * blockDim.x + threadIdx.x;
  if (t >= Bb * Pp * Kk) return;
  int node = t / Kk;
  int p = node % Pp, b = node / Pp;
  int j = idx[t];
  const float* L = &lv[b * 4 * Pp];
  float pxi = L[0 * Pp + p], pyi = L[1 * Pp + p], pzi = L[2 * Pp + p], Ei = L[3 * Pp + p];
  float pxj = L[0 * Pp + j], pyj = L[1 * Pp + j], pzj = L[2 * Pp + j], Ej = L[3 * Pp + j];
  float pti = sqrtf(pxi * pxi + pyi * pyi + EPSF);
  float eti = asinhf(pzi / pti);
  float phi = atan2f(pyi, pxi);
  float ptj = sqrtf(pxj * pxj + pyj * pyj + EPSF);
  float etj = asinhf(pzj / ptj);
  float phj = atan2f(pyj, pxj);
  const float PI = 3.14159265358979323846f;
  const float TWOPI = 6.28318530717958647692f;
  float dd = phi - phj + PI;
  float dphi = dd - floorf(dd / TWOPI) * TWOPI - PI;  // python floored-mod
  float de = eti - etj;
  float lndelta = 0.5f * logf(fmaxf(de * de + dphi * dphi, EPSF));
  float ptmin = fminf(pti, ptj);
  float lnkt = logf(fmaxf(ptmin, EPSF)) + lndelta;
  float lnz = logf(fmaxf(ptmin / fmaxf(pti + ptj, EPSF), EPSF));
  float sx = pxi + pxj, sy = pyi + pyj, sz = pzi + pzj, sE = Ei + Ej;
  float m2 = sE * sE - sx * sx - sy * sy - sz * sz;
  float lnm2 = logf(fmaxf(m2, EPSF));
  float fr[4] = {lndelta, lnkt, lnz, lnm2};
  float er[4];
#pragma unroll
  for (int c = 0; c < 4; ++c) er[c] = fr[c] * ebs[c] + ebb[c];
#pragma unroll
  for (int o = 0; o < EDG; ++o) {
    float v = edge_W[o * 4 + 0] * er[0] + edge_W[o * 4 + 1] * er[1] +
              edge_W[o * 4 + 2] * er[2] + edge_W[o * 4 + 3] * er[3];
    union { unsigned short u; __bf16 h; } z;
    z.h = (__bf16)v;
    e8[(size_t)t * EDG + o] = z.u;
  }
  nulle[t] = (mask[b * Pp + p] && mask[b * Pp + j]) ? 0 : 1;
}

// ---------------- node embed: x0 = node_W @ affine(features) ----------------
__global__ void node_kernel(const float* __restrict__ feat,
                            const float* __restrict__ s0, const float* __restrict__ b0,
                            const float* __restrict__ W, float* __restrict__ x0) {
  int t = blockIdx.x * blockDim.x + threadIdx.x;
  if (t >= Bb * Pp) return;
  int b = t / Pp, p = t % Pp;
  float f[IND];
#pragma unroll
  for (int c = 0; c < IND; ++c) f[c] = feat[(b * IND + c) * Pp + p] * s0[c] + b0[c];
  for (int o = 0; o < NOD; ++o) {
    float v = 0.f;
#pragma unroll
    for (int c = 0; c < IND; ++c) v += W[o * IND + c] * f[c];
    x0[(b * NOD + o) * Pp + p] = v;
  }
}

// -------- prep: W_msg f32 -> zero-padded bf16 [MSG][CMP] --------
__global__ void wprep_kernel(const float* __restrict__ W, int CM, int CMP,
                             unsigned short* __restrict__ out) {
  int t = blockIdx.x * blockDim.x + threadIdx.x;
  if (t >= MSG * CMP) return;
  int m = t / CMP, c = t % CMP;
  union { unsigned short u; __bf16 h; } z;
  z.h = (__bf16)((c < CM) ? W[m * CM + c] : 0.f);
  out[t] = z.u;
}

// ---------------- EdgeConv: 4 waves/block, 8 nodes/wave, WMMA bf16 ----------------
// launch_bounds(threads=128, min 4 waves/SIMD) keeps VGPRs <= 256 (no MSB addressing)
template <int CIN>
__global__ void __launch_bounds__(WAVES * 32, 4)
edgeconv_kernel(const float* __restrict__ x_in,
                const int* __restrict__ idx,
                const int* __restrict__ nulle,
                const unsigned short* __restrict__ e8,
                const unsigned short* __restrict__ wbf,   // prepacked bf16 W_msg
                const float* __restrict__ bn_msg_s, const float* __restrict__ bn_msg_b,
                const float* __restrict__ W_attn,
                const float* __restrict__ W_out,
                const float* __restrict__ bn_out_s, const float* __restrict__ bn_out_b,
                const float* __restrict__ W_sc,
                const float* __restrict__ bn_sc_s, const float* __restrict__ bn_sc_b,
                const float* __restrict__ ls,
                float* __restrict__ x_out) {
  constexpr int CM  = cm_of(CIN);    // 72 / 264
  constexpr int CMP = cmp_of(CIN);   // 96 / 288 (WMMA K multiple)
  constexpr int KSTEPS = CMP / 32;
  constexpr int HP = CMP / 2;        // u32 pairs per column

  extern __shared__ char smem[];
  __bf16* Wh = (__bf16*)smem;                                // [MSG*CMP], block-shared
  char* my = smem + (size_t)MSG * CMP * 2 +
             (size_t)(threadIdx.x >> 5) * perwave_bytes(CIN);
  __bf16* Xh  = (__bf16*)my;                                 // [16*CMP]
  float* mmv  = (float*)(my + (size_t)16 * CMP * 2);         // [MSG*16]
  float* lg   = mmv + MSG * 16;                              // [Hh*16]
  float* catv = lg + Hh * 16;                                // [192]
  float* xi   = catv + 192;                                  // [CIN]
  int*   sj   = (int*)(xi + CIN);                            // [16]
  int*   sn   = sj + 16;                                     // [16]

  const int tid  = threadIdx.x;
  const int wave = tid >> 5;
  const int lane = tid & 31;
  const int nl   = lane & 15;
  const int hi8  = (lane >> 4) * 8;  // 16-bit layout: lanes 16-31 carry K+8

  // stage W once per block (pure u32 copies of prepacked bf16)
  {
    const unsigned* src = reinterpret_cast<const unsigned*>(wbf);
    unsigned* dst = reinterpret_cast<unsigned*>(Wh);
    for (int t = tid; t < MSG * HP; t += WAVES * 32) dst[t] = src[t];
  }
  __syncthreads();

#pragma unroll 1   // one node in flight per wave: keep VGPR pressure bounded
  for (int it = 0; it < NITER; ++it) {
    const int node = blockIdx.x * (WAVES * NITER) + wave * NITER + it;
    const int b = node / Pp;
    const int p = node % Pp;

    if (lane < 16) { sj[lane] = idx[node * Kk + lane]; sn[lane] = nulle[node * Kk + lane]; }
    for (int c = lane; c < CIN; c += 32) xi[c] = x_in[(b * CIN + c) * Pp + p];
    __syncthreads();

    // stage X: Xh[n*CMP + c], written as u32 (2 bf16) pairs
#pragma unroll 1
    for (int t = lane; t < 16 * HP; t += 32) {
      int n = t / HP, c = (t % HP) * 2;
      unsigned pr;
      if (c < CIN) {
        pr = pack_bf(xi[c], xi[c + 1]);
      } else if (c < 2 * CIN) {
        int cc = c - CIN, j = sj[n];
        pr = pack_bf(x_in[(b * CIN + cc) * Pp + j], x_in[(b * CIN + cc + 1) * Pp + j]);
      } else if (c < CM) {
        pr = *reinterpret_cast<const unsigned*>(&e8[((size_t)node * Kk + n) * EDG + (c - 2 * CIN)]);
      } else {
        pr = 0u;
      }
      *reinterpret_cast<unsigned*>(&Xh[n * CMP + c]) = pr;
    }
    __syncthreads();

    union Frag { v16bf v; unsigned u[8]; };
    v8f acc[4];
#pragma unroll
    for (int r = 0; r < 4; ++r)
#pragma unroll
      for (int i = 0; i < 8; ++i) acc[r][i] = 0.f;

#pragma unroll
    for (int ks = 0; ks < KSTEPS; ++ks) {
      const int kk = ks * 32;
      Frag bf;
#pragma unroll
      for (int v = 0; v < 8; ++v) {
        int K0 = ((v < 4) ? 2 * v : 16 + 2 * (v - 4)) + hi8 + kk;  // ISA K-map
        bf.u[v] = *reinterpret_cast<const unsigned*>(&Xh[nl * CMP + K0]);
      }
#pragma unroll
      for (int r = 0; r < 4; ++r) {
        Frag af;
        const int m = r * 16 + nl;
#pragma unroll
        for (int v = 0; v < 8; ++v) {
          int K0 = ((v < 4) ? 2 * v : 16 + 2 * (v - 4)) + hi8 + kk;
          af.u[v] = *reinterpret_cast<const unsigned*>(&Wh[m * CMP + K0]);
        }
        acc[r] = __builtin_amdgcn_wmma_f32_16x16x32_bf16(
            false, af.v, false, bf.v, (short)0, acc[r], false, false);
      }
    }

    // BN + ReLU; C-layout: VGPR i -> M=i (lanes 0-15) / i+8 (16-31)
#pragma unroll
    for (int r = 0; r < 4; ++r)
#pragma unroll
      for (int i = 0; i < 8; ++i) {
        int m = r * 16 + i + hi8;
        float v = acc[r][i] * bn_msg_s[m] + bn_msg_b[m];
        mmv[m * 16 + nl] = v > 0.f ? v : 0.f;
      }
    __syncthreads();

    // attention logits (8 heads x 16 edges), 4 per lane
#pragma unroll
    for (int i = 0; i < 4; ++i) {
      int t = lane * 4 + i;
      int h = t >> 4, n = t & 15;
      float v = 0.f;
#pragma unroll 8
      for (int c = 0; c < MSG; ++c) v += W_attn[h * MSG + c] * mmv[c * 16 + n];
      lg[h * 16 + n] = sn[n] ? -1e9f : v;
    }
    __syncthreads();

    // subset softmax {16,4,8} + aggregation -> cat[192], 6 per lane
#pragma unroll
    for (int i = 0; i < 6; ++i) {
      int t = lane * 6 + i;
      int s = t / 64, w = t % 64, h = w >> 3, d = w & 7;
      int kn = (s == 0) ? 16 : (s == 1 ? 4 : 8);
      float mx = -1e30f;
      for (int k = 0; k < kn; ++k) mx = fmaxf(mx, lg[h * 16 + k]);
      float z = 0.f;
      for (int k = 0; k < kn; ++k) z += __expf(lg[h * 16 + k] - mx);
      float a = 0.f;
      for (int k = 0; k < kn; ++k)
        a += (__expf(lg[h * 16 + k] - mx) / z) * mmv[(h * 8 + d) * 16 + k];
      catv[t] = a;
    }
    __syncthreads();

    // out = BN(W_out @ cat), sc = BN(W_sc @ x_i), x = relu(sc + ls*out)
#pragma unroll
    for (int i = 0; i < 4; ++i) {
      int o = lane * 4 + i;
      float ov = 0.f;
#pragma unroll 8
      for (int c = 0; c < 192; ++c) ov += W_out[o * 192 + c] * catv[c];
      ov = ov * bn_out_s[o] + bn_out_b[o];
      float sv = 0.f;
#pragma unroll 8
      for (int c = 0; c < CIN; ++c) sv += W_sc[o * CIN + c] * xi[c];
      sv = sv * bn_sc_s[o] + bn_sc_b[o];
      float xv = sv + ls[o] * ov;
      x_out[(b * CTX + o) * Pp + p] = xv > 0.f ? xv : 0.f;
    }
    __syncthreads();  // protect Xh/mmv/catv before next iteration overwrites
  }
}

// ---------------- post BN + relu + mask: writes d_out[:,64:192,:] ----------------
__global__ void post_kernel(const float* __restrict__ x,
                            const unsigned char* __restrict__ mask,
                            const float* __restrict__ s, const float* __restrict__ bbv,
                            float* __restrict__ out) {
  int t = blockIdx.x * blockDim.x + threadIdx.x;
  if (t >= Bb * CTX * Pp) return;
  int p = t % Pp;
  int r = t / Pp;
  int o = r % CTX;
  int b = r / CTX;
  float v = x[t] * s[o] + bbv[o];
  v = v > 0.f ? v : 0.f;
  out[(b * (IDD + CTX) + IDD + o) * Pp + p] = v * (mask[b * Pp + p] ? 1.f : 0.f);
}

// ---------------- launch ----------------
extern "C" void kernel_launch(void* const* d_in, const int* in_sizes, int n_in,
                              void* d_out, int out_size, void* d_ws, size_t ws_size,
                              hipStream_t stream) {
  (void)in_sizes; (void)n_in; (void)out_size; (void)ws_size;
  // inputs in setup_inputs() order; params leaves in sorted-pytree order
  const float* points = (const float*)d_in[0];
  const float* feats  = (const float*)d_in[1];
  const float* lv     = (const float*)d_in[2];
  const unsigned char* mask = (const unsigned char*)d_in[3];  // bool (1 byte)
  const float* edge_W    = (const float*)d_in[4];
  const float* edge_bn_b = (const float*)d_in[5];
  const float* edge_bn_s = (const float*)d_in[6];
  const float* id_W      = (const float*)d_in[7];
  const float* id_bn0_b  = (const float*)d_in[8];
  const float* id_bn0_s  = (const float*)d_in[9];
  const float* id_bn1_b  = (const float*)d_in[10];
  const float* id_bn1_s  = (const float*)d_in[11];
  const float* l0_W_attn = (const float*)d_in[12];
  const float* l0_W_msg  = (const float*)d_in[13];
  const float* l0_W_out  = (const float*)d_in[14];
  const float* l0_W_sc   = (const float*)d_in[15];
  const float* l0_bn_msg_b = (const float*)d_in[16];
  const float* l0_bn_msg_s = (const float*)d_in[17];
  const float* l0_bn_out_b = (const float*)d_in[18];
  const float* l0_bn_out_s = (const float*)d_in[19];
  const float* l0_bn_sc_b  = (const float*)d_in[20];
  const float* l0_bn_sc_s  = (const float*)d_in[21];
  const float* l0_ls       = (const float*)d_in[22];
  const float* l1_W_attn = (const float*)d_in[23];
  const float* l1_W_msg  = (const float*)d_in[24];
  const float* l1_W_out  = (const float*)d_in[25];
  const float* l1_W_sc   = (const float*)d_in[26];
  const float* l1_bn_msg_b = (const float*)d_in[27];
  const float* l1_bn_msg_s = (const float*)d_in[28];
  const float* l1_bn_out_b = (const float*)d_in[29];
  const float* l1_bn_out_s = (const float*)d_in[30];
  const float* l1_bn_sc_b  = (const float*)d_in[31];
  const float* l1_bn_sc_s  = (const float*)d_in[32];
  const float* l1_ls       = (const float*)d_in[33];
  const float* node_W    = (const float*)d_in[34];
  const float* node_bn_b = (const float*)d_in[35];
  const float* node_bn_s = (const float*)d_in[36];
  const float* post_bn_b = (const float*)d_in[37];
  const float* post_bn_s = (const float*)d_in[38];

  float* out = (float*)d_out;

  // workspace carve-up
  char* ws = (char*)d_ws;
  size_t off = 0;
  auto carve = [&](size_t bytes) -> void* {
    void* ptr = (void*)(ws + off);
    off = (off + bytes + 255) & ~(size_t)255;
    return ptr;
  };
  int*   idx   = (int*)  carve((size_t)Bb * Pp * Kk * sizeof(int));
  int*   nulle = (int*)  carve((size_t)Bb * Pp * Kk * sizeof(int));
  unsigned short* e8  = (unsigned short*)carve((size_t)Bb * Pp * Kk * EDG * 2);
  float* x0    = (float*)carve((size_t)Bb * NOD * Pp * sizeof(float));
  float* xA    = (float*)carve((size_t)Bb * CTX * Pp * sizeof(float));
  float* xB    = (float*)carve((size_t)Bb * CTX * Pp * sizeof(float));
  unsigned short* wbf0 = (unsigned short*)carve((size_t)MSG * cmp_of(NOD) * 2);
  unsigned short* wbf1 = (unsigned short*)carve((size_t)MSG * cmp_of(CTX) * 2);

  const int NPT = Bb * Pp;
  identity_kernel<<<(NPT + 255) / 256, 256, 0, stream>>>(
      feats, mask, id_bn0_s, id_bn0_b, id_W, id_bn1_s, id_bn1_b, out);
  knn_kernel<<<(NPT + 255) / 256, 256, 0, stream>>>(points, mask, idx);
  edge_kernel<<<(NPT * Kk + 255) / 256, 256, 0, stream>>>(
      lv, mask, idx, edge_bn_s, edge_bn_b, edge_W, e8, nulle);
  node_kernel<<<(NPT + 255) / 256, 256, 0, stream>>>(
      feats, node_bn_s, node_bn_b, node_W, x0);
  wprep_kernel<<<(MSG * cmp_of(NOD) + 255) / 256, 256, 0, stream>>>(
      l0_W_msg, cm_of(NOD), cmp_of(NOD), wbf0);
  wprep_kernel<<<(MSG * cmp_of(CTX) + 255) / 256, 256, 0, stream>>>(
      l1_W_msg, cm_of(CTX), cmp_of(CTX), wbf1);

  const int NBLK = NPT / (WAVES * NITER);  // 512
  edgeconv_kernel<NOD><<<NBLK, WAVES * 32, smem_bytes(NOD), stream>>>(
      x0, idx, nulle, e8, wbf0,
      l0_bn_msg_s, l0_bn_msg_b, l0_W_attn,
      l0_W_out, l0_bn_out_s, l0_bn_out_b,
      l0_W_sc, l0_bn_sc_s, l0_bn_sc_b, l0_ls, xA);

  edgeconv_kernel<CTX><<<NBLK, WAVES * 32, smem_bytes(CTX), stream>>>(
      xA, idx, nulle, e8, wbf1,
      l1_bn_msg_s, l1_bn_msg_b, l1_W_attn,
      l1_W_out, l1_bn_out_s, l1_bn_out_b,
      l1_W_sc, l1_bn_sc_s, l1_bn_sc_b, l1_ls, xB);

  post_kernel<<<(Bb * CTX * Pp + 255) / 256, 256, 0, stream>>>(
      xB, mask, post_bn_s, post_bn_b, out);
}
